// MarGNN_24361054502991
// MI455X (gfx1250) — matrix operand order repaired and checked by reading, not compiled, not run on hardware
//
#include <hip/hip_runtime.h>
#include <hip/hip_bf16.h>

// ---------------- problem constants (from reference) ----------------
constexpr int R_REL  = 3;
constexpr int IN_DIM = 302;
constexpr int KP1    = 320;           // IN_DIM padded to multiple of 32
constexpr int HID    = 128;
constexpr int HEADS  = 4;
constexpr int OUTD   = 64;
constexpr int N0     = 123904;
constexpr int N1     = 11264;
constexpr int BK     = 1024;
constexpr int E1     = 112640;
constexpr int E2     = 10240;
constexpr int HD     = HEADS * HID;   // 512
constexpr int DM     = R_REL * OUTD;  // 192
constexpr float NEG_SLOPE = 0.2f;
constexpr float BN_EPS    = 1e-5f;

typedef __attribute__((ext_vector_type(16))) _Float16 v16h;
typedef __attribute__((ext_vector_type(8)))  _Float16 v8h;
typedef __attribute__((ext_vector_type(8)))  float    v8f;

// ---------------- helpers ----------------
__device__ __forceinline__ unsigned encf(float f) {
    unsigned u = __float_as_uint(f);
    return (u & 0x80000000u) ? ~u : (u | 0x80000000u);
}
__device__ __forceinline__ float decf(unsigned u) {
    return (u & 0x80000000u) ? __uint_as_float(u ^ 0x80000000u)
                             : __uint_as_float(~u);
}
#define ENC_NEG_INF 0x007FFFFFu   // encf(-inf)

// Low 32 bits of a flat LDS pointer == LDS byte offset (aperture lives in addr[63:32]).
__device__ __forceinline__ unsigned lds_addr_of(const void* p) {
    return (unsigned)(unsigned long long)p;
}
__device__ __forceinline__ void wait_async0() {
#if __has_builtin(__builtin_amdgcn_s_wait_asynccnt)
    __builtin_amdgcn_s_wait_asynccnt(0);
#else
    asm volatile("s_wait_asynccnt 0" ::: "memory");
#endif
}
// A-fragment (16-bit A 16x32): two contiguous 8-halves runs -> v16h
__device__ __forceinline__ v16h ld_afrag(const _Float16* __restrict__ p) {
    v8h lo = *(const v8h*)p;
    v8h hi = *(const v8h*)(p + 16);
    return __builtin_shufflevector(lo, hi,
               0, 1, 2, 3, 4, 5, 6, 7, 8, 9, 10, 11, 12, 13, 14, 15);
}
#define WMMA_F16(A_, B_, C_) \
    __builtin_amdgcn_wmma_f32_16x16x32_f16(false, (A_), false, (B_), (short)0, (C_), false, false)

__global__ void fill_f32(float* p, float v, int n) {
    int i = blockIdx.x * blockDim.x + threadIdx.x;
    if (i < n) p[i] = v;
}
__global__ void fill_u32(unsigned* p, unsigned v, int n) {
    int i = blockIdx.x * blockDim.x + threadIdx.x;
    if (i < n) p[i] = v;
}

// ---------------- A-panel producer: gather + f32->f16, zero-padded K ----------------
__global__ void gather_cvt_f16(const float* __restrict__ x, const int* __restrict__ nid,
                               _Float16* __restrict__ xh, int nrows, int Ksrc, int Kpad)
{
    int t = blockIdx.x * blockDim.x + threadIdx.x;
    int groups = Kpad >> 3;
    if (t >= nrows * groups) return;
    int gi = t % groups, i = t / groups;
    int k0 = gi << 3;
    const float* __restrict__ row = x + (long)nid[i] * Ksrc;
    v8h o;
#pragma unroll
    for (int j = 0; j < 8; ++j) {
        int k = k0 + j;
        float v = row[min(k, Ksrc - 1)];          // clamped load, branch-free
        o[j] = (_Float16)((k < Ksrc) ? v : 0.0f);
    }
    *(v8h*)(xh + (long)i * Kpad + k0) = o;
}

// ---------------- B-panel producer: W[K][N] f32 -> Bt[N][Kpad] f16 (zero-padded) ----
__global__ void cvt_w_f16t(const float* __restrict__ W, int Ksrc, int Nn,
                           _Float16* __restrict__ Bt, int Kpad)
{
    int t = blockIdx.x * blockDim.x + threadIdx.x;
    int groups = Kpad >> 3;
    if (t >= Nn * groups) return;
    int gi = t % groups, n = t / groups;
    int k0 = gi << 3;
    v8h o;
#pragma unroll
    for (int j = 0; j < 8; ++j) {
        int k = k0 + j;
        float v = W[(long)min(k, Ksrc - 1) * Nn + n];
        o[j] = (_Float16)((k < Ksrc) ? v : 0.0f);
    }
    *(v8h*)(Bt + (long)n * Kpad + k0) = o;
}

// ---------------- WMMA GEMM: C[M,N] = A[M,K] @ Bt[N,K]^T ----------------
// A:  f16 [M][lda], rows zero-padded to Kpad, lda multiple of 8 (16B rows).
// Bt: f16 [N][ldbt], column-panel transposed + zero-padded (produced above).
// B tile [64n x 32k] f16 double-buffered in LDS via async b128 copies, overlapped
// with the WMMA pipe.  No K tail anywhere (both operands padded).
// block = 256 threads = 8 wave32s; wave grid 4(M) x 2(N); wave tile 32x32;
// block tile 128x64; 4 WMMAs / wave / K-chunk.
constexpr int LDS_K = 32;
constexpr int LDS_N = 64;

__global__ __launch_bounds__(256)
void wmma_gemm(const _Float16* __restrict__ A, int lda,
               const _Float16* __restrict__ Bt, int ldbt,
               float* __restrict__ C, int ldc,
               int M, int N, int Kpad)
{
    __shared__ __align__(128) _Float16 lbt[2][LDS_N * LDS_K];   // [n][k] f16

    const int tid   = threadIdx.x;
    const int lane  = tid & 31;
    const int wave  = tid >> 5;
    const int waveM = wave >> 1;   // 0..3
    const int waveN = wave & 1;    // 0..1

    const int rowBase  = blockIdx.x * 128 + waveM * 32;
    const int colBase0 = blockIdx.y * 64;
    const int nloc0    = waveN * 32 + (lane & 15);
    const int nloc1    = nloc0 + 16;

    // 16-bit A 16x32 fragment: lanes 0-15 row=lane, K runs [0..7],[16..23];
    // lanes 16-31 row=lane-16, K runs [8..15],[24..31].
    // B 32x16 fragment: lane gives col, same K runs as A (at kb).
    const int mrow = lane & 15;
    const int kb   = (lane < 16) ? 0 : 8;
    const int kbb  = (lane < 16) ? 0 : 16;

    const _Float16* __restrict__ Arow0 = A + (long)min(rowBase + mrow,      M - 1) * lda;
    const _Float16* __restrict__ Arow1 = A + (long)min(rowBase + 16 + mrow, M - 1) * lda;

    const unsigned lb_base = lds_addr_of(&lbt[0][0]);

    v8f acc00 = {}, acc01 = {}, acc10 = {}, acc11 = {};

    const int nK = Kpad >> 5;

    // one async b128 per thread covers the whole 64x32 f16 tile (256 x 16B)
    auto stage = [&](int kc, int buf) {
        const int k0   = kc << 5;
        const int n    = tid >> 2;
        const int koff = (tid & 3) << 3;
        const _Float16* g = Bt + (long)(colBase0 + n) * ldbt + k0 + koff;
        unsigned l = lb_base + (unsigned)(buf * (LDS_N * LDS_K * 2) + tid * 16);
        asm volatile("global_load_async_to_lds_b128 %0, %1, off"
                     :: "v"(l), "v"(g) : "memory");
    };

    stage(0, 0);
    wait_async0();
    __syncthreads();

    for (int kc = 0; kc < nK; ++kc) {
        const int k0  = kc << 5;
        const int cur = kc & 1;

        if (kc + 1 < nK) {
            stage(kc + 1, cur ^ 1);                           // overlap with compute
            __builtin_prefetch(Arow0 + ((kc + 1) << 5), 0, 1); // global_prefetch_b8
        }

        v16h a0 = ld_afrag(Arow0 + k0 + kb);
        v16h a1 = ld_afrag(Arow1 + k0 + kb);

        const _Float16* __restrict__ lc = &lbt[cur][0];
        v16h b0 = ld_afrag(lc + nloc0 * LDS_K + kb);
        v16h b1 = ld_afrag(lc + nloc1 * LDS_K + kb);

        acc00 = WMMA_F16(a0, b0, acc00);
        acc01 = WMMA_F16(a0, b1, acc01);
        acc10 = WMMA_F16(a1, b0, acc10);
        acc11 = WMMA_F16(a1, b1, acc11);

        if (kc + 1 < nK) {
            wait_async0();
            __syncthreads();
        }
    }

    // C/D layout: VGPR v -> row v (lanes 0-15) / row 8+v (lanes 16-31); col = lane&15
    const int cfirst = lane & 15;
    const int roff   = (lane < 16) ? 0 : 8;
    const int c0     = colBase0 + waveN * 32 + cfirst;   // call contract: N % 64 == 0
#pragma unroll
    for (int v = 0; v < 8; ++v) {
        int row0 = rowBase + roff + v;
        int row1 = row0 + 16;
        if (row0 < M) {
            C[(long)row0 * ldc + c0]      = acc00[v];
            C[(long)row0 * ldc + c0 + 16] = acc01[v];
        }
        if (row1 < M) {
            C[(long)row1 * ldc + c0]      = acc10[v];
            C[(long)row1 * ldc + c0 + 16] = acc11[v];
        }
    }
}

// ---------------- attention logits: alpha = sum_c h[i,h,c]*a[h,c] ----------------
__global__ void gat_alpha(const float* __restrict__ h,
                          const float* __restrict__ aw_s,
                          const float* __restrict__ aw_d,
                          float* __restrict__ as_out, float* __restrict__ ad_out,
                          int ndst, int H, int C)
{
    const int i    = blockIdx.x;
    const int head = threadIdx.x >> 5;
    const int lane = threadIdx.x & 31;
    const float* __restrict__ row = h + (long)i * H * C + head * C;
    float ss = 0.f, sd = 0.f;
    for (int c = lane; c < C; c += 32) {
        float v = row[c];
        ss += v * aw_s[head * C + c];
        sd += v * aw_d[head * C + c];
    }
#pragma unroll
    for (int m = 16; m; m >>= 1) {
        ss += __shfl_xor(ss, m, 32);
        sd += __shfl_xor(sd, m, 32);
    }
    if (lane == 0) {
        as_out[(long)i * H + head] = ss;
        if (i < ndst) ad_out[(long)i * H + head] = sd;
    }
}

// ---------------- edge softmax (3 passes) + aggregation ----------------
__global__ void edge_logits_max(const int* __restrict__ src, const int* __restrict__ dst,
                                const float* __restrict__ as, const float* __restrict__ ad,
                                unsigned* __restrict__ mmax, int E, int H)
{
    int t = blockIdx.x * blockDim.x + threadIdx.x;
    if (t >= E * H) return;
    int e = t / H, h = t % H;
    float v = as[(long)src[e] * H + h] + ad[(long)dst[e] * H + h];
    v = (v > 0.f) ? v : NEG_SLOPE * v;
    atomicMax(&mmax[(long)dst[e] * H + h], encf(v));
}

__global__ void edge_exp_sum(const int* __restrict__ src, const int* __restrict__ dst,
                             const float* __restrict__ as, const float* __restrict__ ad,
                             const unsigned* __restrict__ mmax,
                             float* __restrict__ ex, float* __restrict__ msum, int E, int H)
{
    int t = blockIdx.x * blockDim.x + threadIdx.x;
    if (t >= E * H) return;
    int e = t / H, h = t % H;
    float v = as[(long)src[e] * H + h] + ad[(long)dst[e] * H + h];
    v = (v > 0.f) ? v : NEG_SLOPE * v;
    float m = decf(mmax[(long)dst[e] * H + h]);
    float x = __expf(v - m);
    ex[t] = x;
    atomicAdd(&msum[(long)dst[e] * H + h], x);
}

__global__ void edge_norm(const int* __restrict__ dst, float* __restrict__ ex,
                          const float* __restrict__ msum, int E, int H)
{
    int t = blockIdx.x * blockDim.x + threadIdx.x;
    if (t >= E * H) return;
    int e = t / H, h = t % H;
    ex[t] = ex[t] / (msum[(long)dst[e] * H + h] + 1e-16f);
}

// thread t = ((e*H)+h)*C + c : agg[dst, h, c] += h_src[src, h, c] * att[e, h]
__global__ void edge_agg(const float* __restrict__ hsrc, const float* __restrict__ att,
                         const int* __restrict__ src, const int* __restrict__ dst,
                         float* __restrict__ agg, int E, int H, int C)
{
    int t = blockIdx.x * blockDim.x + threadIdx.x;
    if (t >= E * H * C) return;
    int c  = t % C;
    int eh = t / C;
    int h  = eh % H;
    int e  = eh / H;
    float v = hsrc[(long)src[e] * H * C + h * C + c] * att[(long)e * H + h];
    atomicAdd(&agg[(long)dst[e] * H * C + h * C + c], v);
}

// ---------------- epilogues ----------------
// mode 0: ELU(BN(x + bias)), mode 1: ReLU(BN(x + bias)); writes f16 (GEMM A operand)
__global__ void bias_bn_act_f16(const float* __restrict__ in, _Float16* __restrict__ out16,
                                const float* __restrict__ bias,
                                const float* __restrict__ g, const float* __restrict__ bb,
                                const float* __restrict__ rm, const float* __restrict__ rv,
                                int n, int ld, int mode)
{
    int i = blockIdx.x * blockDim.x + threadIdx.x;
    if (i >= n) return;
    int c = i % ld;
    float v = in[i] + bias[c];
    v = (v - rm[c]) * g[c] * rsqrtf(rv[c] + BN_EPS) + bb[c];
    if (mode == 0) v = (v > 0.f) ? v : (__expf(v) - 1.f);   // ELU
    else           v = (v > 0.f) ? v : 0.f;                 // ReLU
    out16[i] = (_Float16)v;
}

// fcat16[b, r*OUTD + c] = f16((agg2[b,c] + b2[c]) * thr[0])
__global__ void finalize_rel(const float* __restrict__ agg2, const float* __restrict__ b2,
                             const float* __restrict__ thr, _Float16* __restrict__ fcat16,
                             int r, int n)
{
    int i = blockIdx.x * blockDim.x + threadIdx.x;
    if (i >= n) return;
    int b = i / OUTD, c = i % OUTD;
    fcat16[(long)b * DM + r * OUTD + c] = (_Float16)((agg2[i] + b2[c]) * thr[0]);
}

__global__ void add_bias(float* __restrict__ o, const float* __restrict__ b, int n, int ld) {
    int i = blockIdx.x * blockDim.x + threadIdx.x;
    if (i < n) o[i] += b[i % ld];
}

// ---------------- host launch ----------------
static inline int gsz(long n) { return (int)((n + 255) / 256); }
static inline char* carve(char*& p, size_t bytes) {
    char* q = p;
    p += (bytes + 255) & ~(size_t)255;
    return q;
}

extern "C" void kernel_launch(void* const* d_in, const int* in_sizes, int n_in,
                              void* d_out, int out_size, void* d_ws, size_t ws_size,
                              hipStream_t stream)
{
    const float* x     = (const float*)d_in[0];
    const int*   n_ids = (const int*)  d_in[1];
    const int*   src1  = (const int*)  d_in[2];
    const int*   dst1  = (const int*)  d_in[3];
    const int*   src2  = (const int*)  d_in[4];
    const int*   dst2  = (const int*)  d_in[5];
    const float* W1    = (const float*)d_in[6];
    const float* a1s   = (const float*)d_in[7];
    const float* a1d   = (const float*)d_in[8];
    const float* b1    = (const float*)d_in[9];
    const float* g1    = (const float*)d_in[10];
    const float* bb1   = (const float*)d_in[11];
    const float* rm1   = (const float*)d_in[12];
    const float* rv1   = (const float*)d_in[13];
    const float* W2    = (const float*)d_in[14];
    const float* a2s   = (const float*)d_in[15];
    const float* a2d   = (const float*)d_in[16];
    const float* b2    = (const float*)d_in[17];
    const float* Wm1   = (const float*)d_in[18];
    const float* bm1   = (const float*)d_in[19];
    const float* gm    = (const float*)d_in[20];
    const float* bbm   = (const float*)d_in[21];
    const float* rmm   = (const float*)d_in[22];
    const float* rvm   = (const float*)d_in[23];
    const float* Wm2   = (const float*)d_in[24];
    const float* bm2   = (const float*)d_in[25];
    const float* thr   = (const float*)d_in[26];
    float* out = (float*)d_out;

    // workspace layout (256B-aligned segments)
    char* wp = (char*)d_ws;
    _Float16* xh     = (_Float16*)carve(wp, (size_t)N0 * KP1 * 2);
    float*    h1     = (float*)   carve(wp, (size_t)N0 * HD * 4);
    float*    agg1   = (float*)   carve(wp, (size_t)N1 * HD * 4);
    _Float16* hmidh  = (_Float16*)carve(wp, (size_t)N1 * HD * 2);
    float*    h2     = (float*)   carve(wp, (size_t)N1 * OUTD * 4);
    float*    as1    = (float*)   carve(wp, (size_t)N0 * HEADS * 4);
    float*    ad1    = (float*)   carve(wp, (size_t)N1 * HEADS * 4);
    unsigned* mmax1  = (unsigned*)carve(wp, (size_t)N1 * HEADS * 4);
    float*    msum1  = (float*)   carve(wp, (size_t)N1 * HEADS * 4);
    float*    ex1    = (float*)   carve(wp, (size_t)E1 * HEADS * 4);
    float*    as2    = (float*)   carve(wp, (size_t)N1 * 4);
    float*    ad2    = (float*)   carve(wp, (size_t)BK * 4);
    unsigned* mmax2  = (unsigned*)carve(wp, (size_t)BK * 4);
    float*    msum2  = (float*)   carve(wp, (size_t)BK * 4);
    float*    ex2    = (float*)   carve(wp, (size_t)E2 * 4);
    float*    agg2   = (float*)   carve(wp, (size_t)BK * OUTD * 4);
    _Float16* fcat16 = (_Float16*)carve(wp, (size_t)BK * DM * 2);
    float*    z1     = (float*)   carve(wp, (size_t)BK * DM * 4);
    _Float16* z1h    = (_Float16*)carve(wp, (size_t)BK * DM * 2);
    _Float16* w1t    = (_Float16*)carve(wp, (size_t)HD * KP1 * 2);   // [512][320]
    _Float16* w2t    = (_Float16*)carve(wp, (size_t)OUTD * HD * 2);  // [64][512]
    _Float16* wm1t   = (_Float16*)carve(wp, (size_t)DM * DM * 2);    // [192][192]
    _Float16* wm2t   = (_Float16*)carve(wp, (size_t)OUTD * DM * 2);  // [64][192]

    // MLP weight panels (once per launch)
    cvt_w_f16t<<<gsz(DM * (DM / 8)), 256, 0, stream>>>(Wm1, DM, DM, wm1t, DM);
    cvt_w_f16t<<<gsz(OUTD * (DM / 8)), 256, 0, stream>>>(Wm2, DM, OUTD, wm2t, DM);

    for (int r = 0; r < R_REL; ++r) {
        const int* nid = n_ids + (size_t)r * N0;
        const int* s1  = src1  + (size_t)r * E1;
        const int* t1  = dst1  + (size_t)r * E1;
        const int* s2  = src2  + (size_t)r * E2;
        const int* t2  = dst2  + (size_t)r * E2;

        // operand panels: gathered x -> f16 (padded), W1/W2 -> f16 transposed (padded)
        gather_cvt_f16<<<gsz((long)N0 * (KP1 / 8)), 256, 0, stream>>>(
            x, nid, xh, N0, IN_DIM, KP1);
        cvt_w_f16t<<<gsz(HD * (KP1 / 8)), 256, 0, stream>>>(
            W1 + (size_t)r * IN_DIM * HD, IN_DIM, HD, w1t, KP1);
        cvt_w_f16t<<<gsz(OUTD * (HD / 8)), 256, 0, stream>>>(
            W2 + (size_t)r * HD * OUTD, HD, OUTD, w2t, HD);

        // GEMM1: h1 = xh @ W1[r]   [N0 x 320] @ [320 x 512]
        wmma_gemm<<<dim3(N0 / 128, HD / 64), 256, 0, stream>>>(
            xh, KP1, w1t, KP1, h1, HD, N0, HD, KP1);

        gat_alpha<<<N0, HEADS * 32, 0, stream>>>(
            h1, a1s + (size_t)r * HD, a1d + (size_t)r * HD, as1, ad1, N1, HEADS, HID);

        fill_u32<<<gsz(N1 * HEADS), 256, 0, stream>>>(mmax1, ENC_NEG_INF, N1 * HEADS);
        fill_f32<<<gsz(N1 * HEADS), 256, 0, stream>>>(msum1, 0.f, N1 * HEADS);
        fill_f32<<<gsz((long)N1 * HD), 256, 0, stream>>>(agg1, 0.f, N1 * HD);

        edge_logits_max<<<gsz(E1 * HEADS), 256, 0, stream>>>(s1, t1, as1, ad1, mmax1, E1, HEADS);
        edge_exp_sum  <<<gsz(E1 * HEADS), 256, 0, stream>>>(s1, t1, as1, ad1, mmax1, ex1, msum1, E1, HEADS);
        edge_norm     <<<gsz(E1 * HEADS), 256, 0, stream>>>(t1, ex1, msum1, E1, HEADS);
        edge_agg      <<<gsz((long)E1 * HD), 256, 0, stream>>>(h1, ex1, s1, t1, agg1, E1, HEADS, HID);

        bias_bn_act_f16<<<gsz((long)N1 * HD), 256, 0, stream>>>(
            agg1, hmidh, b1 + (size_t)r * HD, g1 + (size_t)r * HD, bb1 + (size_t)r * HD,
            rm1 + (size_t)r * HD, rv1 + (size_t)r * HD, N1 * HD, HD, /*elu*/0);

        // GEMM2: h2 = hmid @ W2[r]   [N1 x 512] @ [512 x 64]
        wmma_gemm<<<dim3(N1 / 128, 1), 256, 0, stream>>>(
            hmidh, HD, w2t, HD, h2, OUTD, N1, OUTD, HD);

        gat_alpha<<<N1, 32, 0, stream>>>(
            h2, a2s + (size_t)r * OUTD, a2d + (size_t)r * OUTD, as2, ad2, BK, 1, OUTD);

        fill_u32<<<gsz(BK), 256, 0, stream>>>(mmax2, ENC_NEG_INF, BK);
        fill_f32<<<gsz(BK), 256, 0, stream>>>(msum2, 0.f, BK);
        fill_f32<<<gsz(BK * OUTD), 256, 0, stream>>>(agg2, 0.f, BK * OUTD);

        edge_logits_max<<<gsz(E2), 256, 0, stream>>>(s2, t2, as2, ad2, mmax2, E2, 1);
        edge_exp_sum  <<<gsz(E2), 256, 0, stream>>>(s2, t2, as2, ad2, mmax2, ex2, msum2, E2, 1);
        edge_norm     <<<gsz(E2), 256, 0, stream>>>(t2, ex2, msum2, E2, 1);
        edge_agg      <<<gsz((long)E2 * OUTD), 256, 0, stream>>>(h2, ex2, s2, t2, agg2, E2, 1, OUTD);

        finalize_rel<<<gsz(BK * OUTD), 256, 0, stream>>>(agg2, b2 + (size_t)r * OUTD,
                                                         thr + r, fcat16, r, BK * OUTD);
    }

    // MLP: z1 = fcat @ Wm1 ; z1h = relu(BN(z1 + bm1)) ; out = z1h @ Wm2 + bm2
    wmma_gemm<<<dim3(BK / 128, DM / 64), 256, 0, stream>>>(
        fcat16, DM, wm1t, DM, z1, DM, BK, DM, DM);
    bias_bn_act_f16<<<gsz(BK * DM), 256, 0, stream>>>(
        z1, z1h, bm1, gm, bbm, rmm, rvm, BK * DM, DM, /*relu*/1);
    wmma_gemm<<<dim3(BK / 128, 1), 256, 0, stream>>>(
        z1h, DM, wm2t, DM, out, OUTD, BK, OUTD, DM);
    add_bias<<<gsz(BK * OUTD), 256, 0, stream>>>(out, bm2, BK * OUTD, OUTD);
}
